// TripletGAT_19215683682418
// MI455X (gfx1250) — matrix operand order repaired and checked by reading, not compiled, not run on hardware
//
#include <hip/hip_runtime.h>
#include <cstdint>
#include <cstddef>

// ---------------------------------------------------------------------------
// CDNA5 (gfx1250) TripletGAT implementation.
// Dense GEMMs use v_wmma_f32_16x16x32_bf16 (fp32 accumulate).
// ---------------------------------------------------------------------------

typedef __attribute__((ext_vector_type(16))) __bf16 v16bf;
typedef __attribute__((ext_vector_type(8)))  __bf16 v8bf;
typedef __attribute__((ext_vector_type(8)))  float  v8f;

static constexpr int kNPep = 20000, kNMhc = 2000, kNTcr = 50000;
static constexpr int kHid = 256, kHC = 1024;
static constexpr int kEpm = 100000, kEpt = 150000, kEmt = 150000, kB = 50000;

// ---- workspace layout (bytes, all offsets multiples of 32) ----------------
static constexpr size_t OFF_HP   = 0;
static constexpr size_t OFF_HM   = OFF_HP   + (size_t)kNPep * 256 * 4;
static constexpr size_t OFF_HT   = OFF_HM   + (size_t)kNMhc * 256 * 4;
static constexpr size_t OFF_XL   = OFF_HT   + (size_t)kNTcr * 256 * 4;
static constexpr size_t OFF_XR   = OFF_XL   + (size_t)kNPep * 1024 * 2;
static constexpr size_t OFF_AGGM = OFF_XR   + (size_t)kNTcr * 1024 * 2;
static constexpr size_t OFF_AGGT = OFF_AGGM + (size_t)kNMhc * 256 * 4;
static constexpr size_t OFF_ELOG = OFF_AGGT + (size_t)kNTcr * 256 * 4;
static constexpr size_t OFF_SMAX = OFF_ELOG + (size_t)kEpt * 4 * 4;
static constexpr size_t OFF_SDEN = OFF_SMAX + (size_t)kNTcr * 4 * 4;
static constexpr size_t OFF_WPK  = OFF_SDEN + (size_t)kNTcr * 4 * 4;

// packed-weight element offsets (bf16 elements)
static constexpr size_t PK_PROJ0 = 0;
static constexpr size_t PK_PROJ1 = 32768;
static constexpr size_t PK_PROJ2 = 65536;
static constexpr size_t PK_CONV  = 98304;           // 12 x 262144 (Wl/Wr pairs)
static constexpr size_t PK_HEAD  = 98304 + (size_t)12 * 262144;  // 3,244,032

#define DEV __device__ __forceinline__

DEV unsigned ford(float f) {
  unsigned u = __float_as_uint(f);
  return (u & 0x80000000u) ? ~u : (u | 0x80000000u);
}
DEV float forddec(unsigned o) {
  if (o == 0u) return 0.0f;                         // empty segment -> 0
  unsigned u = (o & 0x80000000u) ? (o ^ 0x80000000u) : ~o;
  if ((u & 0x7F800000u) == 0x7F800000u) return 0.0f; // non-finite -> 0
  return __uint_as_float(u);
}
DEV float elu1(float x) { return x > 0.0f ? x : __expf(x) - 1.0f; }

// ---------------------------------------------------------------------------
// Weight packing: fp32 row-major W[K,N] -> bf16 WMMA B-fragment layout.
// flat index i = ((nT*(K/32)+kB)*32 + lane)*16 + j,
//   column n = nT*16 + (lane&15),  K = kB*32 + (lane>>4)*8 + (j&7) + (j>>3)*16
// ---------------------------------------------------------------------------
__global__ __launch_bounds__(256) void pack_w_k(const float* __restrict__ W,
                                                __bf16* __restrict__ dst,
                                                int K, int N) {
  int i = blockIdx.x * 256 + threadIdx.x;
  if (i >= K * N) return;
  int j    = i & 15;
  int lane = (i >> 4) & 31;
  int r    = i >> 9;
  int kBlocks = K >> 5;
  int kB = r % kBlocks;
  int nT = r / kBlocks;
  int n = nT * 16 + (lane & 15);
  int k = kB * 32 + (lane >> 4) * 8 + (j & 7) + (j >> 3) * 16;
  dst[i] = (__bf16)W[(size_t)k * N + n];
}

// ---------------------------------------------------------------------------
// GEMM: C[M,N] = A[M,K](fp32, cvt->bf16) * Bpacked + bias.
// Wave computes a 16x64 tile via 4 accumulators; block = 8 waves = 32x256.
// ---------------------------------------------------------------------------
__global__ __launch_bounds__(256)
void gemm_wmma_bf16(const float* __restrict__ A, const v16bf* __restrict__ Bp,
                    const float* __restrict__ bias, float* __restrict__ Cf,
                    __bf16* __restrict__ Cb, int M, int N, int K) {
  const int lane = threadIdx.x & 31;
  const int wave = threadIdx.x >> 5;
  const int m0 = blockIdx.y * 32 + (wave >> 2) * 16;
  const int n0 = blockIdx.x * 256 + (wave & 3) * 64;
  if (m0 >= M || n0 >= N) return;
  const int half = lane >> 4;
  int mrow = m0 + (lane & 15);
  if (mrow >= M) mrow = M - 1;
  const float* Ar = A + (size_t)mrow * K;
  const int kBlocks = K >> 5;
  const int nT0 = n0 >> 4;

  v8f z = {0.f, 0.f, 0.f, 0.f, 0.f, 0.f, 0.f, 0.f};
  v8f acc0 = z, acc1 = z, acc2 = z, acc3 = z;

  for (int kb = 0; kb < kBlocks; ++kb) {
    const int k0 = (kb << 5) + half * 8;
    float4 f0 = *(const float4*)(Ar + k0);
    float4 f1 = *(const float4*)(Ar + k0 + 4);
    float4 f2 = *(const float4*)(Ar + k0 + 16);
    float4 f3 = *(const float4*)(Ar + k0 + 20);
    v16bf a;
    a[0] = (__bf16)f0.x;  a[1] = (__bf16)f0.y;  a[2] = (__bf16)f0.z;  a[3] = (__bf16)f0.w;
    a[4] = (__bf16)f1.x;  a[5] = (__bf16)f1.y;  a[6] = (__bf16)f1.z;  a[7] = (__bf16)f1.w;
    a[8] = (__bf16)f2.x;  a[9] = (__bf16)f2.y;  a[10] = (__bf16)f2.z; a[11] = (__bf16)f2.w;
    a[12] = (__bf16)f3.x; a[13] = (__bf16)f3.y; a[14] = (__bf16)f3.z; a[15] = (__bf16)f3.w;
    const size_t bb = ((size_t)nT0 * kBlocks + kb) * 32 + lane;
    const size_t st = (size_t)kBlocks * 32;
    v16bf b0 = Bp[bb];
    v16bf b1 = Bp[bb + st];
    v16bf b2 = Bp[bb + 2 * st];
    v16bf b3 = Bp[bb + 3 * st];
    acc0 = __builtin_amdgcn_wmma_f32_16x16x32_bf16(false, a, false, b0, (short)0, acc0, false, false);
    acc1 = __builtin_amdgcn_wmma_f32_16x16x32_bf16(false, a, false, b1, (short)0, acc1, false, false);
    acc2 = __builtin_amdgcn_wmma_f32_16x16x32_bf16(false, a, false, b2, (short)0, acc2, false, false);
    acc3 = __builtin_amdgcn_wmma_f32_16x16x32_bf16(false, a, false, b3, (short)0, acc3, false, false);
  }

  v8f accs[4] = {acc0, acc1, acc2, acc3};
#pragma unroll
  for (int t = 0; t < 4; ++t) {
    const int col = n0 + t * 16 + (lane & 15);
    const float bv = bias[col];
#pragma unroll
    for (int r = 0; r < 8; ++r) {
      const int row = m0 + r + half * 8;
      if (row < M) {
        float v = accs[t][r] + bv;
        if (Cb) Cb[(size_t)row * N + col] = (__bf16)v;
        else    Cf[(size_t)row * N + col] = v;
      }
    }
  }
}

// ---------------------------------------------------------------------------
// Head GEMM: A rows gathered as [hp[ip], hm[im], ht[it]] (K=768), N=256.
// Epilogue fuses ReLU + dot with W2[256] -> atomicAdd into out[row].
// ---------------------------------------------------------------------------
__global__ __launch_bounds__(256)
void head_gemm_wmma(const float* __restrict__ hp, const float* __restrict__ hm,
                    const float* __restrict__ ht,
                    const int* __restrict__ ip, const int* __restrict__ im,
                    const int* __restrict__ it,
                    const v16bf* __restrict__ Bp, const float* __restrict__ b1,
                    const float* __restrict__ W2, float* __restrict__ out, int M) {
  const int lane = threadIdx.x & 31;
  const int wave = threadIdx.x >> 5;
  const int m0 = blockIdx.x * 32 + (wave >> 2) * 16;
  const int n0 = (wave & 3) * 64;
  if (m0 >= M) return;
  const int half = lane >> 4;
  int mrow = m0 + (lane & 15);
  if (mrow >= M) mrow = M - 1;
  const float* sp = hp + (size_t)ip[mrow] * 256;
  const float* sm = hm + (size_t)im[mrow] * 256;
  const float* st = ht + (size_t)it[mrow] * 256;
  const int kBlocks = 24;  // 768/32
  const int nT0 = n0 >> 4;

  v8f z = {0.f, 0.f, 0.f, 0.f, 0.f, 0.f, 0.f, 0.f};
  v8f acc0 = z, acc1 = z, acc2 = z, acc3 = z;

  for (int kb = 0; kb < kBlocks; ++kb) {
    const int sgi = kb >> 3;
    const float* Ar = (sgi == 0 ? sp : (sgi == 1 ? sm : st)) + ((kb & 7) << 5);
    const int k0 = half * 8;
    float4 f0 = *(const float4*)(Ar + k0);
    float4 f1 = *(const float4*)(Ar + k0 + 4);
    float4 f2 = *(const float4*)(Ar + k0 + 16);
    float4 f3 = *(const float4*)(Ar + k0 + 20);
    v16bf a;
    a[0] = (__bf16)f0.x;  a[1] = (__bf16)f0.y;  a[2] = (__bf16)f0.z;  a[3] = (__bf16)f0.w;
    a[4] = (__bf16)f1.x;  a[5] = (__bf16)f1.y;  a[6] = (__bf16)f1.z;  a[7] = (__bf16)f1.w;
    a[8] = (__bf16)f2.x;  a[9] = (__bf16)f2.y;  a[10] = (__bf16)f2.z; a[11] = (__bf16)f2.w;
    a[12] = (__bf16)f3.x; a[13] = (__bf16)f3.y; a[14] = (__bf16)f3.z; a[15] = (__bf16)f3.w;
    const size_t bb = ((size_t)nT0 * kBlocks + kb) * 32 + lane;
    const size_t sstep = (size_t)kBlocks * 32;
    v16bf b0 = Bp[bb];
    v16bf b1 = Bp[bb + sstep];
    v16bf b2 = Bp[bb + 2 * sstep];
    v16bf b3 = Bp[bb + 3 * sstep];
    acc0 = __builtin_amdgcn_wmma_f32_16x16x32_bf16(false, a, false, b0, (short)0, acc0, false, false);
    acc1 = __builtin_amdgcn_wmma_f32_16x16x32_bf16(false, a, false, b1, (short)0, acc1, false, false);
    acc2 = __builtin_amdgcn_wmma_f32_16x16x32_bf16(false, a, false, b2, (short)0, acc2, false, false);
    acc3 = __builtin_amdgcn_wmma_f32_16x16x32_bf16(false, a, false, b3, (short)0, acc3, false, false);
  }

  v8f accs[4] = {acc0, acc1, acc2, acc3};
#pragma unroll
  for (int r = 0; r < 8; ++r) {
    float s = 0.f;
#pragma unroll
    for (int t = 0; t < 4; ++t) {
      const int col = n0 + t * 16 + (lane & 15);
      float v = accs[t][r] + b1[col];
      v = v > 0.f ? v : 0.f;
      s += v * W2[col];
    }
    s += __shfl_xor(s, 1);
    s += __shfl_xor(s, 2);
    s += __shfl_xor(s, 4);
    s += __shfl_xor(s, 8);  // stays within each 16-lane half
    const int row = m0 + r + half * 8;
    if ((lane & 15) == 0 && row < M) atomicAdd(&out[row], s);
  }
}

// ---------------------------------------------------------------------------
// Edge kernels (one wave per edge).
// ---------------------------------------------------------------------------
__global__ __launch_bounds__(256)
void edge_logits_k(const __bf16* __restrict__ xl, const __bf16* __restrict__ xr,
                   const int* __restrict__ src, const int* __restrict__ dst,
                   const float* __restrict__ att, float* __restrict__ elog,
                   unsigned* __restrict__ smax, int E) {
  const int e = (int)((blockIdx.x * 256 + threadIdx.x) >> 5);
  const int lane = threadIdx.x & 31;
  if (e >= E) return;
  const int s = src[e], d = dst[e];
  const __bf16* pl = xl + (size_t)s * kHC + lane * 32;
  const __bf16* pr = xr + (size_t)d * kHC + lane * 32;
  const float* pa = att + lane * 32;
  float part = 0.f;
#pragma unroll
  for (int q = 0; q < 4; ++q) {
    v8bf l8 = ((const v8bf*)pl)[q];
    v8bf r8 = ((const v8bf*)pr)[q];
#pragma unroll
    for (int j = 0; j < 8; ++j) {
      float v = (float)l8[j] + (float)r8[j];
      v = v > 0.f ? v : 0.2f * v;            // leaky_relu(0.2)
      part += v * pa[q * 8 + j];
    }
  }
  part += __shfl_xor(part, 1);
  part += __shfl_xor(part, 2);
  part += __shfl_xor(part, 4);               // reduce 8-lane head groups
  if ((lane & 7) == 0) {
    const int h = lane >> 3;
    elog[(size_t)e * 4 + h] = part;
    atomicMax(&smax[(size_t)d * 4 + h], ford(part));
  }
}

__global__ __launch_bounds__(256)
void edge_exp_k(float* __restrict__ elog, const int* __restrict__ dst,
                const unsigned* __restrict__ smax, float* __restrict__ sden, int E) {
  int t = blockIdx.x * 256 + threadIdx.x;
  if (t >= E * 4) return;
  int e = t >> 2, h = t & 3;
  int d = dst[e];
  float m = forddec(smax[(size_t)d * 4 + h]);
  float ex = __expf(elog[t] - m);
  elog[t] = ex;
  atomicAdd(&sden[(size_t)d * 4 + h], ex);
}

__global__ __launch_bounds__(256)
void edge_agg_k(const float* __restrict__ elog, const float* __restrict__ sden,
                const __bf16* __restrict__ xl, const int* __restrict__ src,
                const int* __restrict__ dst, float* __restrict__ agg, int E) {
  const int e = (int)((blockIdx.x * 256 + threadIdx.x) >> 5);
  const int lane = threadIdx.x & 31;
  if (e >= E) return;
  const int s = src[e], d = dst[e];
  const float a0 = elog[(size_t)e * 4 + 0] / (sden[(size_t)d * 4 + 0] + 1e-16f);
  const float a1 = elog[(size_t)e * 4 + 1] / (sden[(size_t)d * 4 + 1] + 1e-16f);
  const float a2 = elog[(size_t)e * 4 + 2] / (sden[(size_t)d * 4 + 2] + 1e-16f);
  const float a3 = elog[(size_t)e * 4 + 3] / (sden[(size_t)d * 4 + 3] + 1e-16f);
  const __bf16* pl = xl + (size_t)s * kHC;
  float* po = agg + (size_t)d * 256;
#pragma unroll
  for (int q = 0; q < 8; ++q) {
    const int c = lane + q * 32;
    float v = a0 * (float)pl[c] + a1 * (float)pl[256 + c] +
              a2 * (float)pl[512 + c] + a3 * (float)pl[768 + c];
    atomicAdd(&po[c], 0.25f * v);             // fold mean-over-heads
  }
}

// ---------------------------------------------------------------------------
// Node update / init kernels.
// ---------------------------------------------------------------------------
__global__ __launch_bounds__(256) void update_pep_k(float* __restrict__ h, int n) {
  int i = blockIdx.x * 256 + threadIdx.x;
  if (i < n) h[i] = elu1(2.0f * h[i]);
}

__global__ __launch_bounds__(256)
void update_node_k(float* __restrict__ h, const float* __restrict__ agg,
                   const float* __restrict__ b1, const float* __restrict__ b2,
                   float scale, int rows) {
  int i = blockIdx.x * 256 + threadIdx.x;
  int n = rows * 256;
  if (i >= n) return;
  int col = i & 255;
  float o = agg[i] + b1[col] + (b2 ? b2[col] : 0.0f);
  h[i] = elu1(scale * o + h[i]);
}

__global__ __launch_bounds__(256) void fill_u32_k(unsigned* p, unsigned v, int n) {
  int i = blockIdx.x * 256 + threadIdx.x;
  if (i < n) p[i] = v;
}

__global__ __launch_bounds__(256) void out_init_k(float* out, const float* b2, int n) {
  int i = blockIdx.x * 256 + threadIdx.x;
  if (i < n) out[i] = b2[0];
}

// ---------------------------------------------------------------------------
// Host side.
// ---------------------------------------------------------------------------
struct GatP { const float *Wl, *bl, *Wr, *br, *att, *bias; };

extern "C" void kernel_launch(void* const* d_in, const int* in_sizes, int n_in,
                              void* d_out, int out_size, void* d_ws, size_t ws_size,
                              hipStream_t stream) {
  (void)n_in; (void)out_size; (void)ws_size;
  auto F  = [&](int i) { return (const float*)d_in[i]; };
  auto Ii = [&](int i) { return (const int*)d_in[i]; };

  const int *pm_src, *pm_dst, *pt_src, *pt_dst, *mt_src, *mt_dst, *pk_p, *pk_m, *pk_t;
  const float *emb_pep, *emb_mhc, *emb_tcr;
  const float *projW[3], *projB[3];  // pep, mhc, tcr
  GatP gat[2][3];                    // [layer][pm, pt, mt]
  const float *hW1, *hb1, *hW2, *hb2;

  if (in_sizes[0] == 100000) {
    // top-level insertion order for the 9 index arrays
    pm_src = Ii(0); pm_dst = Ii(1); pt_src = Ii(2); pt_dst = Ii(3);
    mt_src = Ii(4); mt_dst = Ii(5); pk_p = Ii(6); pk_m = Ii(7); pk_t = Ii(8);
    if (in_sizes[9] == 2560000) {
      // Layout A: params flattened in insertion order
      emb_pep = F(9); emb_mhc = F(10); emb_tcr = F(11);
      projW[0] = F(12); projB[0] = F(13);
      projW[1] = F(14); projB[1] = F(15);
      projW[2] = F(16); projB[2] = F(17);
      for (int l = 0; l < 2; ++l)
        for (int c = 0; c < 3; ++c) {
          int b = 18 + l * 18 + c * 6;  // conv order pm,pt,mt; inner Wl,bl,Wr,br,att,bias
          gat[l][c] = {F(b + 0), F(b + 1), F(b + 2), F(b + 3), F(b + 4), F(b + 5)};
        }
      hW1 = F(54); hb1 = F(55); hW2 = F(56); hb2 = F(57);
    } else {
      // Layout C: params sorted (jax tree order), top-level positional
      emb_mhc = F(9); emb_pep = F(10); emb_tcr = F(11);
      hW1 = F(12); hb1 = F(13); hW2 = F(14); hb2 = F(15);
      for (int l = 0; l < 2; ++l) {
        int lb = 16 + l * 18;
        int cbase[3] = {lb + 6, lb + 12, lb + 0};  // pm, pt, mt (sorted: mt,pm,pt)
        for (int c = 0; c < 3; ++c) {
          int b = cbase[c];  // sorted inner: Wl,Wr,att,bl,bias,br
          gat[l][c] = {F(b + 0), F(b + 3), F(b + 1), F(b + 5), F(b + 2), F(b + 4)};
        }
      }
      projW[1] = F(52); projB[1] = F(53);
      projW[0] = F(54); projB[0] = F(55);
      projW[2] = F(56); projB[2] = F(57);
    }
  } else {
    // Layout B: fully sorted dict keys at every level
    mt_dst = Ii(0); mt_src = Ii(1); pk_m = Ii(2); pk_p = Ii(3); pk_t = Ii(4);
    emb_mhc = F(5); emb_pep = F(6); emb_tcr = F(7);
    hW1 = F(8); hb1 = F(9); hW2 = F(10); hb2 = F(11);
    for (int l = 0; l < 2; ++l) {
      int lb = 12 + l * 18;
      int cbase[3] = {lb + 6, lb + 12, lb + 0};
      for (int c = 0; c < 3; ++c) {
        int b = cbase[c];
        gat[l][c] = {F(b + 0), F(b + 3), F(b + 1), F(b + 5), F(b + 2), F(b + 4)};
      }
    }
    projW[1] = F(48); projB[1] = F(49);
    projW[0] = F(50); projB[0] = F(51);
    projW[2] = F(52); projB[2] = F(53);
    pm_dst = Ii(54); pm_src = Ii(55); pt_dst = Ii(56); pt_src = Ii(57);
  }

  char* ws = (char*)d_ws;
  float*    hp   = (float*)(ws + OFF_HP);
  float*    hm   = (float*)(ws + OFF_HM);
  float*    ht   = (float*)(ws + OFF_HT);
  __bf16*   xl   = (__bf16*)(ws + OFF_XL);
  __bf16*   xr   = (__bf16*)(ws + OFF_XR);
  float*    aggm = (float*)(ws + OFF_AGGM);
  float*    aggt = (float*)(ws + OFF_AGGT);
  float*    elog = (float*)(ws + OFF_ELOG);
  unsigned* smax = (unsigned*)(ws + OFF_SMAX);
  float*    sden = (float*)(ws + OFF_SDEN);
  __bf16*   wp   = (__bf16*)(ws + OFF_WPK);
  float*    out  = (float*)d_out;

  auto pack = [&](const float* W, size_t off, int K, int N) {
    int n = K * N;
    pack_w_k<<<(n + 255) / 256, 256, 0, stream>>>(W, wp + off, K, N);
  };
  auto gemm = [&](const float* A, size_t pkOff, const float* bias,
                  float* Cf, __bf16* Cb, int M, int N, int K) {
    dim3 g(N / 256, (M + 31) / 32);
    gemm_wmma_bf16<<<g, 256, 0, stream>>>(A, (const v16bf*)(wp + pkOff), bias, Cf, Cb, M, N, K);
  };
  auto fill = [&](void* p, unsigned v, int n) {
    fill_u32_k<<<(n + 255) / 256, 256, 0, stream>>>((unsigned*)p, v, n);
  };

  // ---- pack all GEMM weights into WMMA B-fragment layout ----
  pack(projW[0], PK_PROJ0, 128, 256);
  pack(projW[1], PK_PROJ1, 128, 256);
  pack(projW[2], PK_PROJ2, 128, 256);
  for (int l = 0; l < 2; ++l)
    for (int c = 0; c < 3; ++c) {
      size_t base = PK_CONV + (size_t)(l * 3 + c) * 2 * 262144;
      pack(gat[l][c].Wl, base, 256, 1024);
      pack(gat[l][c].Wr, base + 262144, 256, 1024);
    }
  pack(hW1, PK_HEAD, 768, 256);

  // ---- input projections ----
  gemm(emb_pep, PK_PROJ0, projB[0], hp, nullptr, kNPep, 256, 128);
  gemm(emb_mhc, PK_PROJ1, projB[1], hm, nullptr, kNMhc, 256, 128);
  gemm(emb_tcr, PK_PROJ2, projB[2], ht, nullptr, kNTcr, 256, 128);

  auto conv = [&](const float* hsrc, int Ns, const float* hdst, int Nd,
                  const GatP& p, size_t pkBase,
                  const int* esrc, const int* edst, int E,
                  float* agg, bool zeroAgg) {
    gemm(hsrc, pkBase, p.bl, nullptr, xl, Ns, 1024, 256);
    gemm(hdst, pkBase + 262144, p.br, nullptr, xr, Nd, 1024, 256);
    fill(smax, 0u, Nd * 4);
    fill(sden, 0u, Nd * 4);
    if (zeroAgg) fill(agg, 0u, Nd * 256);
    edge_logits_k<<<(E + 7) / 8, 256, 0, stream>>>(xl, xr, esrc, edst, p.att, elog, smax, E);
    edge_exp_k<<<(E * 4 + 255) / 256, 256, 0, stream>>>(elog, edst, smax, sden, E);
    edge_agg_k<<<(E + 7) / 8, 256, 0, stream>>>(elog, sden, xl, esrc, edst, agg, E);
  };

  for (int l = 0; l < 2; ++l) {
    size_t b0 = PK_CONV + (size_t)(l * 3 + 0) * 2 * 262144;
    size_t b1c = PK_CONV + (size_t)(l * 3 + 1) * 2 * 262144;
    size_t b2c = PK_CONV + (size_t)(l * 3 + 2) * 2 * 262144;
    conv(hp, kNPep, hm, kNMhc, gat[l][0], b0, pm_src, pm_dst, kEpm, aggm, true);
    conv(hp, kNPep, ht, kNTcr, gat[l][1], b1c, pt_src, pt_dst, kEpt, aggt, true);
    conv(hm, kNMhc, ht, kNTcr, gat[l][2], b2c, mt_src, mt_dst, kEmt, aggt, false);
    update_pep_k<<<(kNPep * 256 + 255) / 256, 256, 0, stream>>>(hp, kNPep * 256);
    update_node_k<<<(kNMhc * 256 + 255) / 256, 256, 0, stream>>>(
        hm, aggm, gat[l][0].bias, nullptr, 1.0f, kNMhc);
    update_node_k<<<(kNTcr * 256 + 255) / 256, 256, 0, stream>>>(
        ht, aggt, gat[l][1].bias, gat[l][2].bias, 0.5f, kNTcr);
  }

  // ---- MLP head: gather-concat GEMM + fused ReLU / W2 reduction ----
  out_init_k<<<(kB + 255) / 256, 256, 0, stream>>>(out, hb2, kB);
  head_gemm_wmma<<<(kB + 31) / 32, 256, 0, stream>>>(
      hp, hm, ht, pk_p, pk_m, pk_t, (const v16bf*)(wp + PK_HEAD), hb1, hW2, out, kB);
}
